// Gated_XAtten_Dense_Block_86406152061078
// MI455X (gfx1250) — matrix operand order repaired and checked by reading, not compiled, not run on hardware
//
#include <hip/hip_runtime.h>
#include <math.h>

// ---------------------------------------------------------------------------
// MI455X (gfx1250, wave32) implementation of the gated cross-attention block.
// ~310 GFLOP vs a few hundred MB of traffic at 23.3 TB/s -> compute-bound, so
// every GEMM runs through V_WMMA_F32_16X16X32_BF16 (fp32 accumulate).
// GEMMs stage tiles in LDS via the gfx1250 async global->LDS path (ASYNCcnt)
// when the toolchain exposes the builtins, with a sync fallback otherwise.
// ---------------------------------------------------------------------------

typedef unsigned short u16;
typedef __bf16 bf16_t;
typedef bf16_t bvec16 __attribute__((ext_vector_type(16)));
typedef float  fvec8  __attribute__((ext_vector_type(8)));
typedef int    gv4i   __attribute__((vector_size(16)));  // matches builtin sig

#define DEV static __device__ __forceinline__

#if defined(__gfx1250__) &&                                         \
    __has_builtin(__builtin_amdgcn_global_load_async_to_lds_b128) && \
    __has_builtin(__builtin_amdgcn_s_wait_asynccnt)
#define HAVE_ASYNC_LDS 1
#else
#define HAVE_ASYNC_LDS 0
#endif

constexpr int kB = 4, kN = 1024, kC = 1024, kH = 16, kD = 64, kMid = 4096;
constexpr int kRows = kB * kN;     // 4096
constexpr int k2Mid = 2 * kMid;    // 8192

DEV u16 f2bf(float f) {
  union { float f; unsigned u; } c; c.f = f;
  unsigned u = c.u;
  return (u16)((u + 0x7FFFu + ((u >> 16) & 1u)) >> 16);  // RNE
}
DEV float bf2f(u16 h) {
  union { unsigned u; float f; } c; c.u = ((unsigned)h) << 16;
  return c.f;
}

union FragB16 { bvec16 v; uint4 q[2]; };

// A-matrix 16x32 bf16 fragment: lane<16 -> K 0..15,
// lane>=16 -> K 16..31 (kb is 0 or 8).  Two aligned 16B loads per lane.
DEV bvec16 load_frag_a(const u16* p, int kb) {
  FragB16 f;
  f.q[0] = *(const uint4*)(p + 2 * kb);
  f.q[1] = *(const uint4*)(p + 2 * kb + 8);
  return f.v;
}
// B-matrix 32x16 bf16 fragment: lanes 0-15 hold K=0..15, lanes 16-31 K=16..31
// -> 32 contiguous bytes per lane (caller folds hi*16 into p).
DEV bvec16 load_frag_b(const u16* p) {
  FragB16 f;
  f.q[0] = *(const uint4*)(p);
  f.q[1] = *(const uint4*)(p + 8);
  return f.v;
}
DEV fvec8 vzero8() { fvec8 z = {0.f,0.f,0.f,0.f,0.f,0.f,0.f,0.f}; return z; }

DEV fvec8 wmma_bf16(bvec16 a, bvec16 b, fvec8 c) {
  return __builtin_amdgcn_wmma_f32_16x16x32_bf16(
      /*neg_a=*/false, a, /*neg_b=*/false, b,
      /*c_mod=*/(short)0, c, /*reuse_a=*/false, /*reuse_b=*/false);
}

// 16B global -> LDS copy: async DMA on gfx1250 (ASYNCcnt) or sync fallback.
DEV void cp_b128_to_lds(u16* lds, const u16* g) {
#if HAVE_ASYNC_LDS
  __builtin_amdgcn_global_load_async_to_lds_b128(
      (__attribute__((address_space(1))) gv4i*)(u16*)g,
      (__attribute__((address_space(3))) gv4i*)lds, 0, 0);
#else
  *(uint4*)lds = *(const uint4*)g;
#endif
}
DEV void wait_async_all() {
#if HAVE_ASYNC_LDS
  __builtin_amdgcn_s_wait_asynccnt(0);  // s_wait_asynccnt 0
#endif
}

// ---------------------------------------------------------------------------
// fp32 [K,N] -> bf16 transposed [N,K]  (weight prep so B operand is Bt)
// ---------------------------------------------------------------------------
__global__ void transpose_bf16_kernel(const float* __restrict__ src,
                                      u16* __restrict__ dst, int K, int N) {
  size_t i = (size_t)blockIdx.x * 256 + threadIdx.x;
  if (i >= (size_t)K * N) return;
  int n = (int)(i % N);
  int k = (int)(i / N);
  dst[(size_t)n * K + k] = f2bf(src[i]);
}

// ---------------------------------------------------------------------------
// LayerNorm over C=1024, one row per block (256 thr = 8 waves).
// ---------------------------------------------------------------------------
__global__ __launch_bounds__(256) void ln_kernel(
    const float* __restrict__ x, const float* __restrict__ w,
    const float* __restrict__ b, float* __restrict__ of32,
    u16* __restrict__ obf) {
  const int row = blockIdx.x, tid = threadIdx.x;
  const float* xr = x + (size_t)row * kC;
  float vals[4], s = 0.f, ss = 0.f;
#pragma unroll
  for (int i = 0; i < 4; ++i) {
    float v = xr[tid + i * 256];
    vals[i] = v; s += v; ss += v * v;
  }
#pragma unroll
  for (int o = 16; o >= 1; o >>= 1) {
    s  += __shfl_xor(s,  o, 32);
    ss += __shfl_xor(ss, o, 32);
  }
  __shared__ float red[16];
  int wid = tid >> 5, lane = tid & 31;
  if (lane == 0) { red[wid] = s; red[8 + wid] = ss; }
  __syncthreads();
  s = 0.f; ss = 0.f;
#pragma unroll
  for (int i = 0; i < 8; ++i) { s += red[i]; ss += red[8 + i]; }
  float mean = s * (1.0f / kC);
  float var  = ss * (1.0f / kC) - mean * mean;
  float rstd = rsqrtf(var + 1e-6f);
#pragma unroll
  for (int i = 0; i < 4; ++i) {
    int c = tid + i * 256;
    float o = (vals[i] - mean) * rstd * w[c] + b[c];
    if (of32) of32[(size_t)row * kC + c] = o;
    if (obf)  obf[(size_t)row * kC + c]  = f2bf(o);
  }
}

// ---------------------------------------------------------------------------
// LDS-staged, double-buffered bf16 WMMA GEMM: C[M,N] = A[M,K] * Bt[N,K]^T.
// Block 256 thr = 8 waves; block tile 128x128; wave tile 32x64 (8 wmma/k-step).
// Tiles staged via async global->LDS (ASYNCcnt) with ds_load_b128 fragment
// reads; LDS rows padded to 40 elements (80B) to spread banks.
// ---------------------------------------------------------------------------
constexpr int GP = 40;  // padded LDS row stride (elements)

template <bool OUT_BF16>
__global__ __launch_bounds__(256) void gemm_kernel(
    const u16* __restrict__ A, const u16* __restrict__ Bt,
    void* __restrict__ Cout, int M, int N, int K) {
  __shared__ __align__(16) u16 sA[2][128 * GP];  // 2 x 10 KB
  __shared__ __align__(16) u16 sB[2][128 * GP];  // 2 x 10 KB

  const int tid = threadIdx.x;
  const int wid = tid >> 5, lane = tid & 31;
  const int lm = lane & 15, hi = lane >> 4;
  const int kb = hi ? 8 : 0;
  const int m0 = (wid & 3) * 32;   // wave tile origin inside block tile
  const int n0 = (wid >> 2) * 64;
  const int mb = blockIdx.y * 128, nb = blockIdx.x * 128;

  fvec8 acc[2][4];
#pragma unroll
  for (int i = 0; i < 2; ++i)
#pragma unroll
    for (int j = 0; j < 4; ++j) acc[i][j] = vzero8();

  // Cooperative tile loader: 128 rows x 32 cols bf16 per matrix per k-step;
  // 512 16B-chunks per matrix -> 2 chunks/thread/matrix.
  auto issue = [&](int buf, int kk) {
#pragma unroll
    for (int c = tid; c < 512; c += 256) {
      int row = c >> 2, col = (c & 3) * 8;
      cp_b128_to_lds(&sA[buf][row * GP + col],
                     A + (size_t)(mb + row) * K + kk + col);
      cp_b128_to_lds(&sB[buf][row * GP + col],
                     Bt + (size_t)(nb + row) * K + kk + col);
    }
  };

  const int nk = K >> 5;
  issue(0, 0);
  int buf = 0;
  for (int k = 0; k < nk; ++k) {
    wait_async_all();   // this wave's async batch for `buf` has landed
    __syncthreads();    // every wave's batch has landed; prior reads done
    if (k + 1 < nk) issue(buf ^ 1, (k + 1) << 5);  // overlap DMA with math

    const u16* pa0 = &sA[buf][(m0 + lm) * GP];
    const u16* pa1 = &sA[buf][(m0 + 16 + lm) * GP];
    bvec16 a0 = load_frag_a(pa0, kb);
    bvec16 a1 = load_frag_a(pa1, kb);
    bvec16 bf[4];
#pragma unroll
    for (int j = 0; j < 4; ++j)
      bf[j] = load_frag_b(&sB[buf][(n0 + j * 16 + lm) * GP + hi * 16]);
#pragma unroll
    for (int j = 0; j < 4; ++j) {
      acc[0][j] = wmma_bf16(a0, bf[j], acc[0][j]);
      acc[1][j] = wmma_bf16(a1, bf[j], acc[1][j]);
    }
    buf ^= 1;
  }

#pragma unroll
  for (int i = 0; i < 2; ++i)
#pragma unroll
    for (int j = 0; j < 4; ++j) {
      int n = nb + n0 + j * 16 + lm;
#pragma unroll
      for (int r = 0; r < 8; ++r) {
        int m = mb + m0 + i * 16 + hi * 8 + r;  // C/D: VGPR r -> row r + 8*hi
        float v = acc[i][j][r];
        if (OUT_BF16) ((u16*)Cout)[(size_t)m * N + n]   = f2bf(v);
        else          ((float*)Cout)[(size_t)m * N + n] = v;
      }
    }
}

// ---------------------------------------------------------------------------
// RoPE + head split: fp32 [B,N,*] slice -> bf16 [B,H,N,64] (optional scale).
// ---------------------------------------------------------------------------
__global__ void rope_heads_kernel(const float* __restrict__ src,
                                  const float* __restrict__ pos,
                                  u16* __restrict__ dst, int ldsrc, int coloff,
                                  float scale) {
  size_t i = (size_t)blockIdx.x * 256 + threadIdx.x;
  if (i >= (size_t)kRows * kC) return;
  int c = (int)(i % kC);
  size_t bn = i / kC;
  int n = (int)(bn % kN), b = (int)(bn / kN);
  int h = c >> 6, t = c & 63;
  const float* s = src + bn * ldsrc + coloff;
  float p = pos[n * kD + t];
  float v = s[c];
  float rot = (t < 32) ? -s[c + 32] : s[c - 32];
  float o = (v * __cosf(p) + rot * __sinf(p)) * scale;
  dst[(((size_t)(b * kH + h)) * kN + n) * kD + t] = f2bf(o);
}

// V: fp32 [B,N,*] slice -> bf16 transposed [B,H,64,N] so WMMA B fragments of
// V are contiguous loads along the key dimension.
__global__ void v_heads_t_kernel(const float* __restrict__ src,
                                 u16* __restrict__ dst, int ldsrc, int coloff) {
  size_t i = (size_t)blockIdx.x * 256 + threadIdx.x;
  if (i >= (size_t)kRows * kC) return;
  int c = (int)(i % kC);
  size_t bn = i / kC;
  int n = (int)(bn % kN), b = (int)(bn / kN);
  int h = c >> 6, t = c & 63;
  dst[(((size_t)(b * kH + h)) * kD + t) * kN + n] = f2bf(src[bn * ldsrc + coloff + c]);
}

// ---------------------------------------------------------------------------
// Flash attention (bf16 WMMA, fp32 online softmax).
// grid = (N/128, B*H); 256 thr = 8 waves; each wave: 16 query rows.
// Q: [B,H,N,64] bf16 (scaled+roped), K: [B,H,N,64] bf16 (roped),
// V: [B,H,64,N] bf16.  Out: merged bf16 [B,N,C].
// ---------------------------------------------------------------------------
__global__ __launch_bounds__(256) void flash_attn_kernel(
    const u16* __restrict__ Q, const u16* __restrict__ Kc,
    const u16* __restrict__ Vt, const unsigned char* __restrict__ mask,
    u16* __restrict__ Out) {
  const int tid = threadIdx.x;
  const int w = tid >> 5, lane = tid & 31;
  const int lm = lane & 15, hi = lane >> 4, kb = hi ? 8 : 0;
  const int bh = blockIdx.y;
  const int b = bh / kH, h = bh % kH;
  const int qbase = blockIdx.x * 128 + w * 16;

  __shared__ __align__(32) u16 ldsP[8][16 * 32];  // per-wave P tile (16x32)

  const u16* Qrow = Q + ((size_t)bh * kN + qbase + lm) * kD;
  bvec16 aq0 = load_frag_a(Qrow, kb);        // d = 0..31
  bvec16 aq1 = load_frag_a(Qrow + 32, kb);   // d = 32..63

  float mrow[8], lrow[8];
  fvec8 oacc[4];
#pragma unroll
  for (int r = 0; r < 8; ++r) { mrow[r] = -1e30f; lrow[r] = 0.f; }
#pragma unroll
  for (int f = 0; f < 4; ++f) oacc[f] = vzero8();

  for (int kt = 0; kt < kN; kt += 32) {
#pragma unroll
    for (int sub = 0; sub < 2; ++sub) {
      int key = kt + sub * 16 + lm;  // this lane's key column
      const u16* Krow = Kc + ((size_t)bh * kN + key) * kD + hi * 16;
      bvec16 bk0 = load_frag_b(Krow);
      bvec16 bk1 = load_frag_b(Krow + 32);
      fvec8 s = vzero8();
      s = wmma_bf16(aq0, bk0, s);
      s = wmma_bf16(aq1, bk1, s);
      float keym = (mask && !mask[b * kN + key]) ? -1e30f : 0.0f;
#pragma unroll
      for (int r = 0; r < 8; ++r) {
        float sv = s[r] + keym;
        // row reductions stay inside a 16-lane half (xor masks 1,2,4,8)
        float mx = sv;
        mx = fmaxf(mx, __shfl_xor(mx, 1, 32));
        mx = fmaxf(mx, __shfl_xor(mx, 2, 32));
        mx = fmaxf(mx, __shfl_xor(mx, 4, 32));
        mx = fmaxf(mx, __shfl_xor(mx, 8, 32));
        float mnew  = fmaxf(mrow[r], mx);
        float alpha = __expf(mrow[r] - mnew);
        float p     = __expf(sv - mnew);
        float rs = p;
        rs += __shfl_xor(rs, 1, 32);
        rs += __shfl_xor(rs, 2, 32);
        rs += __shfl_xor(rs, 4, 32);
        rs += __shfl_xor(rs, 8, 32);
        lrow[r] = lrow[r] * alpha + rs;
        mrow[r] = mnew;
        oacc[0][r] *= alpha; oacc[1][r] *= alpha;
        oacc[2][r] *= alpha; oacc[3][r] *= alpha;
        ldsP[w][(hi * 8 + r) * 32 + sub * 16 + lm] = f2bf(p);
      }
    }
    // Re-read P as an A-matrix fragment (ds_load_b128, same-wave ordering).
    bvec16 pa = load_frag_a(&ldsP[w][lm * 32], kb);
#pragma unroll
    for (int f = 0; f < 4; ++f) {
      const u16* Vrow = Vt + ((size_t)bh * kD + f * 16 + lm) * kN + kt + hi * 16;
      bvec16 bv = load_frag_b(Vrow);
      oacc[f] = wmma_bf16(pa, bv, oacc[f]);
    }
  }
#pragma unroll
  for (int f = 0; f < 4; ++f) {
    int t = f * 16 + lm;
#pragma unroll
    for (int r = 0; r < 8; ++r) {
      int m = qbase + hi * 8 + r;
      float inv = 1.0f / fmaxf(lrow[r], 1e-30f);
      Out[((size_t)(b * kN + m)) * kC + h * kD + t] = f2bf(oacc[f][r] * inv);
    }
  }
}

// ---------------------------------------------------------------------------
// Fused SwiGLU + RMSNorm: h bf16 [rows, 8192] -> out bf16 [rows, 4096].
// ---------------------------------------------------------------------------
__global__ __launch_bounds__(256) void swiglu_rms_kernel(
    const u16* __restrict__ h, const float* __restrict__ g,
    u16* __restrict__ out) {
  const int row = blockIdx.x, tid = threadIdx.x;
  __shared__ float ubuf[kMid];  // 16 KB of the 320 KB WGP LDS
  __shared__ float red[8];
  const u16* hr = h + (size_t)row * k2Mid;
  float ss = 0.f;
  for (int i = tid; i < kMid; i += 256) {
    float x1 = bf2f(hr[i]);
    float gt = bf2f(hr[kMid + i]);
    float u  = gt / (1.0f + __expf(-gt)) * x1;  // silu(gate) * x1
    ubuf[i] = u;
    ss += u * u;
  }
#pragma unroll
  for (int o = 16; o >= 1; o >>= 1) ss += __shfl_xor(ss, o, 32);
  int wid = tid >> 5, lane = tid & 31;
  if (lane == 0) red[wid] = ss;
  __syncthreads();
  ss = 0.f;
#pragma unroll
  for (int i = 0; i < 8; ++i) ss += red[i];
  float nrm   = sqrtf(ss) * 0.015625f;  // * mid^-0.5
  float scale = 1.0f / fmaxf(nrm, 1e-8f);
  for (int i = tid; i < kMid; i += 256)
    out[(size_t)row * kMid + i] = f2bf(ubuf[i] * scale * g[i]);
}

// out = base + (gate ? tanh(gate[0]) : 1) * y
__global__ void residual_kernel(const float* __restrict__ base,
                                const float* __restrict__ y,
                                const float* __restrict__ gate,
                                float* __restrict__ out, int n) {
  int i = blockIdx.x * 256 + threadIdx.x;
  if (i >= n) return;
  float gv = gate ? tanhf(gate[0]) : 1.0f;
  out[i] = base[i] + gv * y[i];
}

// ---------------------------------------------------------------------------
extern "C" void kernel_launch(void* const* d_in, const int* in_sizes, int n_in,
                              void* d_out, int out_size, void* d_ws,
                              size_t ws_size, hipStream_t stream) {
  (void)in_sizes; (void)n_in; (void)out_size; (void)ws_size;
  const float* q   = (const float*)d_in[0];
  const float* kin = (const float*)d_in[1];
  const float* vin = (const float*)d_in[2];
  const float* pos = (const float*)d_in[3];
  const unsigned char* mask = (const unsigned char*)d_in[4];
  // LN params: n1q n1k n1v n1p n2 n3 n3p n4 at indices 6..21 (w,b pairs)
  const float* n1q_w = (const float*)d_in[6];  const float* n1q_b = (const float*)d_in[7];
  const float* n1k_w = (const float*)d_in[8];  const float* n1k_b = (const float*)d_in[9];
  const float* n1v_w = (const float*)d_in[10]; const float* n1v_b = (const float*)d_in[11];
  const float* n1p_w = (const float*)d_in[12]; const float* n1p_b = (const float*)d_in[13];
  const float* n2_w  = (const float*)d_in[14]; const float* n2_b  = (const float*)d_in[15];
  const float* n3_w  = (const float*)d_in[16]; const float* n3_b  = (const float*)d_in[17];
  const float* n3p_w = (const float*)d_in[18]; const float* n3p_b = (const float*)d_in[19];
  const float* n4_w  = (const float*)d_in[20]; const float* n4_b  = (const float*)d_in[21];
  const float* mlp1_g = (const float*)d_in[29];
  const float* mlp2_g = (const float*)d_in[32];
  const float* attn_gate = (const float*)d_in[34];
  const float* ff_gate   = (const float*)d_in[35];
  float* outp = (float*)d_out;

  char* wsp = (char*)d_ws;
  size_t off = 0;
  auto alloc = [&](size_t bytes) -> void* {
    void* p = wsp + off;
    off += (bytes + 255) & ~(size_t)255;
    return p;
  };
  const size_t RC = (size_t)kRows * kC;  // 4M elements

  // bf16 transposed weights (~64 MB)
  u16* tw_caq   = (u16*)alloc((size_t)kC * kC * 2);
  u16* tw_cak   = (u16*)alloc((size_t)kC * kC * 2);
  u16* tw_cav   = (u16*)alloc((size_t)kC * kC * 2);
  u16* tw_capr  = (u16*)alloc((size_t)kC * kC * 2);
  u16* tw_saqkv = (u16*)alloc((size_t)kC * 3 * kC * 2);
  u16* tw_sapr  = (u16*)alloc((size_t)kC * kC * 2);
  u16* tw_m1w1  = (u16*)alloc((size_t)kC * k2Mid * 2);
  u16* tw_m1w2  = (u16*)alloc((size_t)kMid * kC * 2);
  u16* tw_m2w1  = (u16*)alloc((size_t)kC * k2Mid * 2);
  u16* tw_m2w2  = (u16*)alloc((size_t)kMid * kC * 2);
  // activations
  u16*  lnbuf = (u16*)alloc(RC * 2);                        // 8 MB
  void* bigr  = alloc((size_t)kRows * k2Mid * 2);           // 64 MB, shared:
  float* projf = (float*)bigr;                              //   q/k/v or qkv fp32
  u16*   hbuf  = (u16*)bigr;                                //   MLP hidden bf16
  u16* Qh   = (u16*)alloc(RC * 2);
  u16* Kh   = (u16*)alloc(RC * 2);
  u16* Vth  = (u16*)alloc(RC * 2);
  u16* Obuf = (u16*)alloc(RC * 2);
  float* yb  = (float*)alloc(RC * 4);
  float* x1f = (float*)alloc(RC * 4);
  float* x2f = (float*)alloc(RC * 4);
  u16* ubuf = (u16*)alloc((size_t)kRows * kMid * 2);        // 32 MB

  dim3 blk(256);
  const float scale = 0.125f;  // d^-0.5, folded into Q after RoPE (linear)
  hipStream_t s = stream;

  auto tr = [&](int idx, u16* dst, int K, int N) {
    size_t tot = (size_t)K * N;
    transpose_bf16_kernel<<<dim3((unsigned)((tot + 255) / 256)), blk, 0, s>>>(
        (const float*)d_in[idx], dst, K, N);
  };
  auto gemm = [&](const u16* A, const u16* Bt, void* Cp, int M, int N, int K,
                  bool obf) {
    dim3 g(N / 128, M / 128);
    if (obf) gemm_kernel<true><<<g, blk, 0, s>>>(A, Bt, Cp, M, N, K);
    else     gemm_kernel<false><<<g, blk, 0, s>>>(A, Bt, Cp, M, N, K);
  };
  auto ln = [&](const float* x, const float* w, const float* b, float* of,
                u16* ob) { ln_kernel<<<kRows, blk, 0, s>>>(x, w, b, of, ob); };
  const unsigned nEW = (unsigned)(RC / 256);  // elementwise grid (16384)

  // ---- weight prep -------------------------------------------------------
  tr(22, tw_caq, kC, kC);     tr(23, tw_cak, kC, kC);
  tr(24, tw_cav, kC, kC);     tr(25, tw_capr, kC, kC);
  tr(26, tw_saqkv, kC, 3 * kC); tr(27, tw_sapr, kC, kC);
  tr(28, tw_m1w1, kC, k2Mid); tr(30, tw_m1w2, kMid, kC);
  tr(31, tw_m2w1, kC, k2Mid); tr(33, tw_m2w2, kMid, kC);

  // ---- gated cross-attention --------------------------------------------
  ln(q, n1q_w, n1q_b, nullptr, lnbuf);
  gemm(lnbuf, tw_caq, projf, kRows, kC, kC, false);
  ln(kin, n1k_w, n1k_b, nullptr, lnbuf);
  gemm(lnbuf, tw_cak, projf + RC, kRows, kC, kC, false);
  ln(vin, n1v_w, n1v_b, nullptr, lnbuf);
  gemm(lnbuf, tw_cav, projf + 2 * RC, kRows, kC, kC, false);
  rope_heads_kernel<<<nEW, blk, 0, s>>>(projf, pos, Qh, kC, 0, scale);
  rope_heads_kernel<<<nEW, blk, 0, s>>>(projf + RC, pos, Kh, kC, 0, 1.0f);
  v_heads_t_kernel<<<nEW, blk, 0, s>>>(projf + 2 * RC, Vth, kC, 0);
  flash_attn_kernel<<<dim3(kN / 128, kB * kH), blk, 0, s>>>(Qh, Kh, Vth, mask, Obuf);
  gemm(Obuf, tw_capr, yb, kRows, kC, kC, false);
  residual_kernel<<<nEW, blk, 0, s>>>(q, yb, attn_gate, x1f, (int)RC);
  ln(x1f, n1p_w, n1p_b, x2f, nullptr);       // x = ln(x)  -> x2f

  // ---- gated FFN (SwiGLU MLP 1) -----------------------------------------
  ln(x2f, n2_w, n2_b, nullptr, lnbuf);
  gemm(lnbuf, tw_m1w1, hbuf, kRows, k2Mid, kC, true);
  swiglu_rms_kernel<<<kRows, blk, 0, s>>>(hbuf, mlp1_g, ubuf);
  gemm(ubuf, tw_m1w2, yb, kRows, kC, kMid, false);
  residual_kernel<<<nEW, blk, 0, s>>>(x2f, yb, ff_gate, x1f, (int)RC);

  // ---- self-attention ----------------------------------------------------
  ln(x1f, n3_w, n3_b, nullptr, lnbuf);
  gemm(lnbuf, tw_saqkv, projf, kRows, 3 * kC, kC, false);
  rope_heads_kernel<<<nEW, blk, 0, s>>>(projf, pos, Qh, 3 * kC, 0, scale);
  rope_heads_kernel<<<nEW, blk, 0, s>>>(projf, pos, Kh, 3 * kC, kC, 1.0f);
  v_heads_t_kernel<<<nEW, blk, 0, s>>>(projf, Vth, 3 * kC, 2 * kC);
  flash_attn_kernel<<<dim3(kN / 128, kB * kH), blk, 0, s>>>(Qh, Kh, Vth, nullptr, Obuf);
  gemm(Obuf, tw_sapr, yb, kRows, kC, kC, false);
  residual_kernel<<<nEW, blk, 0, s>>>(x1f, yb, nullptr, x2f, (int)RC);
  ln(x2f, n3p_w, n3p_b, x1f, nullptr);       // x = ln(x)  -> x1f

  // ---- SwiGLU MLP 2 + final residual ------------------------------------
  ln(x1f, n4_w, n4_b, nullptr, lnbuf);
  gemm(lnbuf, tw_m2w1, hbuf, kRows, k2Mid, kC, true);
  swiglu_rms_kernel<<<kRows, blk, 0, s>>>(hbuf, mlp2_g, ubuf);
  gemm(ubuf, tw_m2w2, yb, kRows, kC, kMid, false);
  residual_kernel<<<nEW, blk, 0, s>>>(x1f, yb, nullptr, outp, (int)RC);
}
